// BulkFormer_59270548685258
// MI455X (gfx1250) — compile-verified, matrix operand order checked by reading, not run on hardware
//
#include <hip/hip_runtime.h>
#include <hip/hip_bf16.h>

// ---------------------------------------------------------------------------
// BulkFormer forward for MI455X (gfx1250, wave32).
// Matmuls run through v_wmma_f32_16x16x32_f16 with f16 operands staged into
// LDS via async global->LDS b128 copies (double buffered), fp32 accumulate.
// GEMM epilogue optionally emits a fused f16 copy of C to feed the next GEMM
// without a separate conversion pass over HBM.
// ---------------------------------------------------------------------------

typedef _Float16 f16;
typedef __attribute__((ext_vector_type(16))) _Float16 v16h;
typedef __attribute__((ext_vector_type(8)))  float    v8f;

constexpr int cB = 2, cG = 20000, cD = 256, cEMB = 320, cDEG = 20;
constexpr int cBINS = 10, cHEADS = 2, cDH = 32, cMF = 128;
constexpr int cE = cG * cDEG;          // 400000 edges
constexpr long cBG = (long)cB * cG;    // 40000 rows
constexpr int cNB = cG / cBINS;        // 2000 rows per bin
constexpr int cN2 = 32768;             // bitonic padding (>= cG)

// GEMM tiling: block tile 128x128, 8 waves, wave tile 32x64 (8 WMMA / K-step)
constexpr int BM = 128, BN = 128, BK = 32;

// ------------------------------- device utils ------------------------------

__device__ inline float wave_sum(float v) {
  #pragma unroll
  for (int o = 16; o > 0; o >>= 1) v += __shfl_xor(v, o, 32);
  return v;
}
__device__ inline float wave_max(float v) {
  #pragma unroll
  for (int o = 16; o > 0; o >>= 1) v = fmaxf(v, __shfl_xor(v, o, 32));
  return v;
}
__device__ inline float gelu_f(float x) {
  const float c = 0.7978845608028654f;
  float t = tanhf(c * (x + 0.044715f * x * x * x));
  return 0.5f * x * (1.0f + t);
}
// monotonic float<->uint encode for atomicMax on floats
__device__ inline unsigned fenc(float f) {
  unsigned u = __float_as_uint(f);
  return (u & 0x80000000u) ? ~u : (u | 0x80000000u);
}
__device__ inline float fdec(unsigned u) {
  return __uint_as_float((u & 0x80000000u) ? (u & 0x7fffffffu) : ~u);
}

// CDNA5 async global->LDS copy (ASYNCcnt-tracked), see CDNA5 ISA.
__device__ inline void async_b128(unsigned lds_off, const void* gaddr) {
  asm volatile("global_load_async_to_lds_b128 %0, %1, off"
               :: "v"(lds_off), "v"(gaddr) : "memory");
}
__device__ inline void wait_async_4() {
  asm volatile("s_wait_asynccnt 4" ::: "memory");
}
__device__ inline void wait_async_0() {
  asm volatile("s_wait_asynccnt 0" ::: "memory");
}

// ------------------------------- GEMM (WMMA) --------------------------------
// C[M,N] = act(A[M,K] * Bt[N,K]^T + bias[N] + res[M,N]); optional f16 copy C16.
// A,Bt f16 row-major with leading dims lda/ldb (elements), K multiple of 32.
template <int ACT>
__global__ __launch_bounds__(256)
void k_gemm(const f16* __restrict__ A, int lda,
            const f16* __restrict__ Bt, int ldb,
            float* __restrict__ C, int ldc,
            f16* __restrict__ C16,
            const float* __restrict__ bias,
            const float* __restrict__ res, int ldr,
            int Mdim, int Ndim, int Kdim) {
  extern __shared__ f16 smem[];
  f16* As = smem;                         // [2][BM*BK]
  f16* Bs = smem + 2 * BM * BK;           // [2][BN*BK]
  const int tid = threadIdx.x;
  const int bM = blockIdx.x * BM;
  const int bN = blockIdx.y * BN;
  const unsigned ldsB0 = (unsigned)(2 * BM * BK * sizeof(f16));

  auto stage = [&](int k0, int buf) {
    // A tile: 128 rows x 32 halves = 512 x 16B transfers, 2 per thread
    #pragma unroll
    for (int t = 0; t < 2; ++t) {
      int idx = tid + t * 256;
      int row = idx >> 2, seg = idx & 3;
      int gr = bM + row; if (gr >= Mdim) gr = Mdim - 1;
      const f16* g = A + (long)gr * lda + k0 + seg * 8;
      unsigned l = (unsigned)(((buf * BM * BK) + row * BK + seg * 8) * sizeof(f16));
      async_b128(l, (const void*)g);
    }
    // B tile: 128 rows x 32 halves = 512 x 16B transfers, 2 per thread
    #pragma unroll
    for (int t = 0; t < 2; ++t) {
      int idx = tid + t * 256;
      int row = idx >> 2, seg = idx & 3;
      int gn = bN + row; if (gn >= Ndim) gn = Ndim - 1;
      const f16* g = Bt + (long)gn * ldb + k0 + seg * 8;
      unsigned l = ldsB0 + (unsigned)(((buf * BN * BK) + row * BK + seg * 8) * sizeof(f16));
      async_b128(l, (const void*)g);
    }
  };

  const int lane = tid & 31, wid = tid >> 5;
  const int wm = (wid & 3) * 32;   // 4 waves along M (2 m-tiles each)
  const int wn = (wid >> 2) * 64;  // 2 waves along N (4 n-tiles each)
  v8f acc[2][4];
  #pragma unroll
  for (int i = 0; i < 2; ++i)
    #pragma unroll
    for (int j = 0; j < 4; ++j)
      #pragma unroll
      for (int r = 0; r < 8; ++r) acc[i][j][r] = 0.0f;

  const int nk = Kdim / BK;
  stage(0, 0);
  for (int kt = 0; kt < nk; ++kt) {
    const int buf = kt & 1;
    if (kt + 1 < nk) stage((kt + 1) * BK, buf ^ 1);
    if (kt + 1 < nk) wait_async_4(); else wait_async_0();
    __syncthreads();

    const f16* Abuf = As + buf * BM * BK;
    const f16* Bbuf = Bs + buf * BN * BK;
    const int r = lane & 15;
    const int k0 = (lane >> 4) << 3;  // 0 or 8 (CDNA5 16-bit A/B frag layout)
    v16h af[2], bf[4];
    #pragma unroll
    for (int i = 0; i < 2; ++i) {
      const f16* pa = Abuf + (wm + i * 16 + r) * BK + k0;
      #pragma unroll
      for (int u = 0; u < 8; ++u) { af[i][u] = pa[u]; af[i][8 + u] = pa[16 + u]; }
    }
    #pragma unroll
    for (int j = 0; j < 4; ++j) {
      const f16* pb = Bbuf + (wn + j * 16 + r) * BK + k0;
      #pragma unroll
      for (int u = 0; u < 8; ++u) { bf[j][u] = pb[u]; bf[j][8 + u] = pb[16 + u]; }
    }
    #pragma unroll
    for (int i = 0; i < 2; ++i)
      #pragma unroll
      for (int j = 0; j < 4; ++j)
        acc[i][j] = __builtin_amdgcn_wmma_f32_16x16x32_f16(
            false, af[i], false, bf[j], (short)0, acc[i][j], false, false);
    __syncthreads();
  }

  // epilogue: lane holds (row = rb+r, col) of each 16x16 tile
  const int col = lane & 15;
  const int rb = (lane >> 4) << 3;
  #pragma unroll
  for (int i = 0; i < 2; ++i)
    #pragma unroll
    for (int j = 0; j < 4; ++j)
      #pragma unroll
      for (int rI = 0; rI < 8; ++rI) {
        int gr = bM + wm + i * 16 + rb + rI;
        int gc = bN + wn + j * 16 + col;
        if (gr < Mdim && gc < Ndim) {
          float v = acc[i][j][rI];
          if (bias) v += bias[gc];
          if (res)  v += res[(long)gr * ldr + gc];
          if (ACT == 1) v = v > 0.f ? v : 0.f;
          else if (ACT == 2) v = gelu_f(v);
          C[(long)gr * ldc + gc] = v;
          if (C16) C16[(long)gr * ldc + gc] = (f16)v;
        }
      }
}

// ------------------------- small / elementwise kernels ----------------------

// f32 -> f16, optional transpose, pad K dim with zeros.
__global__ void k_cvt(const float* __restrict__ in, int ldin, f16* __restrict__ out,
                      long rows, int kdim, int kpad, int trans) {
  long t = blockIdx.x * 256L + threadIdx.x;
  long tot = rows * (long)kpad;
  if (t >= tot) return;
  long r = t / kpad; int k = (int)(t % kpad);
  float v = 0.f;
  if (k < kdim) v = trans ? in[(long)k * ldin + r] : in[r * ldin + k];
  out[t] = (f16)v;
}

__global__ void k_fill_f32(float* p, float v, long n) {
  long t = blockIdx.x * 256L + threadIdx.x; if (t < n) p[t] = v;
}
__global__ void k_fill_u32(unsigned* p, unsigned v, long n) {
  long t = blockIdx.x * 256L + threadIdx.x; if (t < n) p[t] = v;
}

// wave-per-row LayerNorm; also emits f16 copy for the next WMMA GEMM
__global__ void k_ln(const float* __restrict__ x, const float* __restrict__ gg,
                     const float* __restrict__ bb, float* __restrict__ y,
                     f16* __restrict__ y16, long rows, int dim) {
  long w = (blockIdx.x * 256L + threadIdx.x) >> 5;
  int lane = threadIdx.x & 31;
  if (w >= rows) return;
  const float* xr = x + w * dim;
  float s = 0.f;
  for (int i = lane; i < dim; i += 32) s += xr[i];
  s = wave_sum(s) / dim;
  float v = 0.f;
  for (int i = lane; i < dim; i += 32) { float d = xr[i] - s; v += d * d; }
  v = wave_sum(v) / dim;
  float rs = rsqrtf(v + 1e-5f);
  for (int i = lane; i < dim; i += 32) {
    float o = (xr[i] - s) * rs * gg[i] + bb[i];
    y[w * dim + i] = o;
    if (y16) y16[w * dim + i] = (f16)o;
  }
}

// h = expr(x) + gene_tok + ae_latent  (+ fused f16 copy)
__global__ void k_embed(const float* __restrict__ x, const float* __restrict__ invf,
                        const float* __restrict__ gt, const float* __restrict__ lat,
                        float* __restrict__ h, f16* __restrict__ h16) {
  long t = blockIdx.x * 256L + threadIdx.x;
  if (t >= cBG * (long)cD) return;
  int d = (int)(t & (cD - 1));
  long bg = t >> 8;
  int g = (int)(bg % cG), b = (int)(bg / cG);
  float xv = x[(long)b * cG + g];
  float e = (d < cD / 2) ? sinf(xv * invf[d]) : cosf(xv * invf[d - cD / 2]);
  if (xv == -10.0f) e = 0.f;
  float o = e + gt[(long)g * cD + d] + lat[(long)b * cD + d];
  h[t] = o;
  if (h16) h16[t] = (f16)o;
}

// GATv2: wave per (batch, edge): e = leaky_relu(xl[src]+xr[dst]) . att; seg-max
__global__ void k_gat_edge(const float* __restrict__ xl, const float* __restrict__ xr,
                           const int* __restrict__ src, const int* __restrict__ dst,
                           const float* __restrict__ att, float* __restrict__ eb,
                           unsigned* __restrict__ mm) {
  long w = (blockIdx.x * 256L + threadIdx.x) >> 5;
  int lane = threadIdx.x & 31;
  if (w >= (long)cB * cE) return;
  int b = (int)(w / cE), e = (int)(w % cE);
  int s0 = src[e], d0 = dst[e];
  const float* pl = xl + ((long)b * cG + s0) * cD;
  const float* pr = xr + ((long)b * cG + d0) * cD;
  float acc = 0.f;
  for (int i = lane; i < cD; i += 32) {
    float v = pl[i] + pr[i];
    v = v > 0.f ? v : 0.2f * v;
    acc += v * att[i];
  }
  acc = wave_sum(acc);
  if (lane == 0) {
    eb[w] = acc;
    atomicMax(&mm[(long)b * cG + d0], fenc(acc));
  }
}

__global__ void k_gat_soft(float* __restrict__ eb, const unsigned* __restrict__ mm,
                           const int* __restrict__ dst, float* __restrict__ den) {
  long t = blockIdx.x * 256L + threadIdx.x;
  if (t >= (long)cB * cE) return;
  int b = (int)(t / cE), e = (int)(t % cE);
  int d0 = dst[e];
  float m = fdec(mm[(long)b * cG + d0]);
  if (!isfinite(m)) m = 0.f;
  float ex = expf(eb[t] - m);
  eb[t] = ex;
  atomicAdd(&den[(long)b * cG + d0], ex);
}

__global__ void k_gat_agg(const float* __restrict__ xl, const float* __restrict__ eb,
                          const float* __restrict__ den, const int* __restrict__ src,
                          const int* __restrict__ dst, float* __restrict__ gx) {
  long w = (blockIdx.x * 256L + threadIdx.x) >> 5;
  int lane = threadIdx.x & 31;
  if (w >= (long)cB * cE) return;
  int b = (int)(w / cE), e = (int)(w % cE);
  int s0 = src[e], d0 = dst[e];
  float alpha = eb[w] / (den[(long)b * cG + d0] + 1e-16f);
  const float* pl = xl + ((long)b * cG + s0) * cD;
  float* po = gx + ((long)b * cG + d0) * cD;
  for (int i = lane; i < cD; i += 32) atomicAdd(&po[i], pl[i] * alpha);
}

__global__ void k_gatmix(const float* __restrict__ xn, const float* __restrict__ gx,
                         const float* __restrict__ bias, float* __restrict__ out,
                         long rows, int dim) {
  long t = blockIdx.x * 256L + threadIdx.x;
  if (t >= rows * dim) return;
  out[t] = xn[t] + gx[t] + bias[(int)(t % dim)];
}

// GEMV: out[row] = act(dot(A[row,:K], w) + bias);  act: 0=id 1=relu 3=1/(v+eps)
__global__ void k_gemv(const float* __restrict__ A, int lda, long sA,
                       const float* __restrict__ w, long sw,
                       const float* __restrict__ bias, float* __restrict__ out,
                       long so, int rows, int K, int act) {
  long gw = (blockIdx.x * 256L + threadIdx.x) >> 5;
  int lane = threadIdx.x & 31;
  int bz = blockIdx.z;
  if (gw >= rows) return;
  const float* a = A + (long)bz * sA + gw * lda;
  const float* ww = w + (long)bz * sw;
  float s = 0.f;
  for (int i = lane; i < K; i += 32) s += a[i] * ww[i];
  s = wave_sum(s);
  if (lane == 0) {
    if (bias) s += bias[0];
    if (act == 1) s = fmaxf(s, 0.f);
    else if (act == 3) s = 1.0f / (s + 1e-6f);
    out[(long)bz * so + gw] = s;
  }
}

// --- bitonic sort (descending) of (score, idx) per batch, padded to cN2 ---
__global__ void k_sortinit(const float* __restrict__ sc, float* __restrict__ keys,
                           int* __restrict__ vals) {
  long t = blockIdx.x * 256L + threadIdx.x;
  if (t >= (long)cB * cN2) return;
  int b = (int)(t >> 15), i = (int)(t & (cN2 - 1));
  keys[t] = (i < cG) ? sc[(long)b * cG + i] : -3.4e38f;
  vals[t] = i;
}
__global__ void k_bitonic(float* __restrict__ keys, int* __restrict__ vals,
                          int kk, int jj) {
  long t = blockIdx.x * 256L + threadIdx.x;
  if (t >= (long)cB * cN2) return;
  int b = (int)(t >> 15), i = (int)(t & (cN2 - 1));
  int p = i ^ jj;
  if (p <= i) return;
  long bi = (long)b * cN2;
  float ki = keys[bi + i], kp = keys[bi + p];
  bool desc = ((i & kk) == 0);
  if (desc ? (ki < kp) : (ki > kp)) {
    keys[bi + i] = kp; keys[bi + p] = ki;
    int v = vals[bi + i]; vals[bi + i] = vals[bi + p]; vals[bi + p] = v;
  }
}
__global__ void k_copyord(const int* __restrict__ vals, int* __restrict__ ord) {
  long t = blockIdx.x * 256L + threadIdx.x;
  if (t >= (long)cB * cG) return;
  ord[t] = vals[(t / cG) * cN2 + (t % cG)];
}
__global__ void k_invperm(const int* __restrict__ ord, int* __restrict__ inv) {
  long t = blockIdx.x * 256L + threadIdx.x;
  if (t >= (long)cB * cG) return;
  long b = t / cG; int i = (int)(t % cG);
  inv[b * cG + ord[t]] = i;
}
__global__ void k_gather(const float* __restrict__ in, const int* __restrict__ idx,
                         float* __restrict__ out) {
  long t = blockIdx.x * 256L + threadIdx.x;
  if (t >= cBG * (long)cD) return;
  int d = (int)(t & (cD - 1));
  long bg = t >> 8;
  long b = bg / cG; int i = (int)(bg % cG);
  int j = idx[b * cG + i];
  out[t] = in[(b * cG + j) * cD + d];
}

// pack/unpack strided batch slices into contiguous rows
__global__ void k_pack(const float* __restrict__ in, long bstride, int nrows,
                       int nb, int dim, float* __restrict__ out) {
  long t = blockIdx.x * 256L + threadIdx.x;
  long tot = (long)nb * nrows * dim;
  if (t >= tot) return;
  int d = (int)(t % dim); long r = t / dim;
  long b = r / nrows; long rr = r % nrows;
  out[t] = in[b * bstride + rr * dim + d];
}
__global__ void k_unpack(const float* __restrict__ in, float* __restrict__ out,
                         long bstride, int nrows, int nb, int dim) {
  long t = blockIdx.x * 256L + threadIdx.x;
  long tot = (long)nb * nrows * dim;
  if (t >= tot) return;
  int d = (int)(t % dim); long r = t / dim;
  long b = r / nrows; long rr = r % nrows;
  out[b * bstride + rr * dim + d] = in[t];
}

// --- FAVOR+ helpers ---
// out[r,m] = M^-0.5 * (exp(s*dd[r,m] - 0.5*s^2*||q_r||^2 - mx) + 1e-4)
__global__ void k_favor(const float* __restrict__ dd, const float* __restrict__ qh,
                        int ldq, float* __restrict__ out, f16* __restrict__ out16,
                        long rows, int nper, const float* __restrict__ mxk) {
  long w = (blockIdx.x * 256L + threadIdx.x) >> 5;
  int lane = threadIdx.x & 31;
  if (w >= rows) return;
  const float sc = 0.42044820763f;   // 32^-0.25
  const float hs2 = 0.08838834765f;  // 0.5 * 32^-0.5
  const float mn = 0.08838834765f;   // 128^-0.5
  float qv = qh[w * (long)ldq + lane];
  float diag = hs2 * wave_sum(qv * qv);
  float mx;
  if (mxk) mx = mxk[w / nper];
  else {
    float m = -3.4e38f;
    for (int j = lane; j < cMF; j += 32) m = fmaxf(m, sc * dd[w * cMF + j]);
    mx = wave_max(m);
  }
  for (int j = lane; j < cMF; j += 32) {
    float e = sc * dd[w * cMF + j] - diag - mx;
    float o = mn * (expf(e) + 1e-4f);
    out[w * cMF + j] = o;
    if (out16) out16[w * cMF + j] = (f16)o;
  }
}
__global__ void k_rowmax(const float* __restrict__ dd, float* __restrict__ rmax,
                         long rows) {
  long w = (blockIdx.x * 256L + threadIdx.x) >> 5;
  int lane = threadIdx.x & 31;
  if (w >= rows) return;
  const float sc = 0.42044820763f;
  float m = -3.4e38f;
  for (int j = lane; j < cMF; j += 32) m = fmaxf(m, sc * dd[w * cMF + j]);
  m = wave_max(m);
  if (lane == 0) rmax[w] = m;
}
__global__ void k_bmax(const float* __restrict__ rmax, float* __restrict__ mxk, int n) {
  __shared__ float sh[256];
  float m = -3.4e38f;
  for (int i = threadIdx.x; i < n; i += 256)
    m = fmaxf(m, rmax[(long)blockIdx.x * n + i]);
  sh[threadIdx.x] = m;
  __syncthreads();
  for (int s = 128; s > 0; s >>= 1) {
    if (threadIdx.x < s) sh[threadIdx.x] = fmaxf(sh[threadIdx.x], sh[threadIdx.x + s]);
    __syncthreads();
  }
  if (threadIdx.x == 0) mxk[blockIdx.x] = sh[0];
}
__global__ void k_colsum(const float* __restrict__ kp, float* __restrict__ ks,
                         int n, int nb) {
  int t = blockIdx.x * 256 + threadIdx.x;
  if (t >= nb * cMF) return;
  int b = t >> 7, m = t & (cMF - 1);
  const float* p = kp + (long)b * n * cMF + m;
  float s = 0.f;
  for (int i = 0; i < n; ++i) {
    if ((i & 63) == 0) __builtin_prefetch(p + (long)(i + 128) * cMF, 0, 0);
    s += p[(long)i * cMF];
  }
  ks[t] = s;
}
// O[r,c] *= z[r], plus fused f16 copy (feeds the Wo projection GEMM)
__global__ void k_scalerows(float* __restrict__ O, f16* __restrict__ O16, int ld,
                            int cols, const float* __restrict__ z, long rows) {
  long t = blockIdx.x * 256L + threadIdx.x;
  if (t >= rows * cols) return;
  long r = t / cols; int c = (int)(t % cols);
  float v = O[r * (long)ld + c] * z[r];
  O[r * (long)ld + c] = v;
  O16[r * (long)ld + c] = (f16)v;
}

// ------------------------------- host side ---------------------------------

static inline dim3 gthreads(long total) {
  return dim3((unsigned)((total + 255) / 256));
}

static void gemm(hipStream_t s, const f16* A, int lda, const f16* Bt, int ldb,
                 float* C, int ldc, f16* C16, const float* bias,
                 const float* res, int ldr, int M, int N, int K, int act) {
  dim3 g((unsigned)((M + BM - 1) / BM), (unsigned)((N + BN - 1) / BN), 1);
  size_t sh = (size_t)2 * (BM * BK + BN * BK) * sizeof(f16);
  if (act == 0)      k_gemm<0><<<g, 256, sh, s>>>(A, lda, Bt, ldb, C, ldc, C16, bias, res, ldr, M, N, K);
  else if (act == 1) k_gemm<1><<<g, 256, sh, s>>>(A, lda, Bt, ldb, C, ldc, C16, bias, res, ldr, M, N, K);
  else               k_gemm<2><<<g, 256, sh, s>>>(A, lda, Bt, ldb, C, ldc, C16, bias, res, ldr, M, N, K);
}
static void cvt(hipStream_t s, const float* in, int ldin, f16* out, long rows,
                int kdim, int kpad, int trans) {
  long tot = rows * (long)kpad;
  k_cvt<<<gthreads(tot), 256, 0, s>>>(in, ldin, out, rows, kdim, kpad, trans);
}

// ------- parameter tree (jax flattens dicts in sorted-key order) -------
struct Dense { const float* b; const float* w; };
struct PerfP {
  Dense ff1, ff2, k, o, q, v;
  const float *ln1b, *ln1g, *ln2b, *ln2g, *proj;
};
struct GatP { const float *att, *bias; Dense wl, wr; };
struct BlockP { PerfP bins; PerfP full; GatP gat; const float *lnb, *lng; Dense score; };

static PerfP parse_perf(void* const* in, int& i) {
  PerfP p;
  p.ff1.b = (const float*)in[i++]; p.ff1.w = (const float*)in[i++];
  p.ff2.b = (const float*)in[i++]; p.ff2.w = (const float*)in[i++];
  p.k.b = (const float*)in[i++];   p.k.w = (const float*)in[i++];
  p.ln1b = (const float*)in[i++];  p.ln1g = (const float*)in[i++];
  p.ln2b = (const float*)in[i++];  p.ln2g = (const float*)in[i++];
  p.o.b = (const float*)in[i++];   p.o.w = (const float*)in[i++];
  p.proj = (const float*)in[i++];
  p.q.b = (const float*)in[i++];   p.q.w = (const float*)in[i++];
  p.v.b = (const float*)in[i++];   p.v.w = (const float*)in[i++];
  return p;
}
static PerfP perf_at_bin(const PerfP& s, int bi) {
  PerfP p;
  p.ff1.b = s.ff1.b + (long)bi * 1024;      p.ff1.w = s.ff1.w + (long)bi * 256 * 1024;
  p.ff2.b = s.ff2.b + (long)bi * 256;       p.ff2.w = s.ff2.w + (long)bi * 1024 * 256;
  p.k.b = s.k.b + (long)bi * 64;            p.k.w = s.k.w + (long)bi * 256 * 64;
  p.ln1b = s.ln1b + (long)bi * 256;         p.ln1g = s.ln1g + (long)bi * 256;
  p.ln2b = s.ln2b + (long)bi * 256;         p.ln2g = s.ln2g + (long)bi * 256;
  p.o.b = s.o.b + (long)bi * 256;           p.o.w = s.o.w + (long)bi * 64 * 256;
  p.proj = s.proj + (long)bi * 128 * 32;
  p.q.b = s.q.b + (long)bi * 64;            p.q.w = s.q.w + (long)bi * 256 * 64;
  p.v.b = s.v.b + (long)bi * 64;            p.v.w = s.v.w + (long)bi * 256 * 64;
  return p;
}

struct Pool {
  float *H, *XN, *XL, *XR, *GX, *HS, *T1024, *PX;
  float *Q, *K_, *V_, *O, *DD, *QP, *KP, *Z, *KS, *RMAX, *MXK;
  float *EB, *DEN, *SC, *KEYS, *KV;
  f16 *XN16, *T1024h, *W16, *PROJ16, *Q16, *K16, *O16, *QP16, *KT16, *VT16, *KVT16;
  unsigned* MM;
  int *VALS, *ORD, *INV;
  float *PA, *PB;  // aliases of XR / GX during performer phases
};

// one pre-norm Performer layer (FAVOR+ attention + GELU FFN) on packed slices
static void run_performer(hipStream_t s, const PerfP& p, float* base, long bstride,
                          int n, int nb, Pool& P) {
  const long TR = (long)nb * n;
  const int npad = (n + 31) & ~31;

  k_pack<<<gthreads(TR * cD), 256, 0, s>>>(base, bstride, n, nb, cD, P.PX);
  k_ln<<<gthreads(TR * 32), 256, 0, s>>>(P.PX, p.ln1g, p.ln1b, P.XN, P.XN16, TR, cD);

  // q,k,v projections (Bt = W^T f16); f16 copies of q,k fused in the epilogue
  cvt(s, p.q.w, 64, P.W16, 64, 256, 256, 1);
  gemm(s, P.XN16, 256, P.W16, 256, P.Q, 64, P.Q16, p.q.b, nullptr, 0, (int)TR, 64, 256, 0);
  cvt(s, p.k.w, 64, P.W16, 64, 256, 256, 1);
  gemm(s, P.XN16, 256, P.W16, 256, P.K_, 64, P.K16, p.k.b, nullptr, 0, (int)TR, 64, 256, 0);
  cvt(s, p.v.w, 64, P.W16, 64, 256, 256, 1);
  gemm(s, P.XN16, 256, P.W16, 256, P.V_, 64, nullptr, p.v.b, nullptr, 0, (int)TR, 64, 256, 0);

  cvt(s, p.proj, 32, P.PROJ16, 128, 32, 32, 0);

  for (int h = 0; h < cHEADS; ++h) {
    // dd_q = q_h @ proj^T ; qp = favor(dd_q, rowwise max)  (f16 copy fused)
    gemm(s, P.Q16 + h * 32, 64, P.PROJ16, 32, P.DD, cMF, nullptr, nullptr, nullptr, 0,
         (int)TR, cMF, 32, 0);
    k_favor<<<gthreads(TR * 32), 256, 0, s>>>(P.DD, P.Q + h * 32, 64, P.QP, P.QP16,
                                              TR, n, (const float*)nullptr);
    // dd_k ; kp = favor(dd_k, per-batch global max)
    gemm(s, P.K16 + h * 32, 64, P.PROJ16, 32, P.DD, cMF, nullptr, nullptr, nullptr, 0,
         (int)TR, cMF, 32, 0);
    k_rowmax<<<gthreads(TR * 32), 256, 0, s>>>(P.DD, P.RMAX, TR);
    k_bmax<<<dim3(nb), 256, 0, s>>>(P.RMAX, P.MXK, n);
    k_favor<<<gthreads(TR * 32), 256, 0, s>>>(P.DD, P.K_ + h * 32, 64, P.KP,
                                              (f16*)nullptr, TR, n, P.MXK);
    // ksum, z
    k_colsum<<<gthreads((long)nb * cMF), 256, 0, s>>>(P.KP, P.KS, n, nb);
    {
      dim3 gz((unsigned)(((long)n * 32 + 255) / 256), 1, (unsigned)nb);
      k_gemv<<<gz, 256, 0, s>>>(P.QP, cMF, (long)n * cMF, P.KS, cMF, nullptr,
                                P.Z, n, n, cMF, 3);
    }
    for (int b = 0; b < nb; ++b) {
      // kv = kp^T @ v_h  (K padded to multiple of 32, zero filled)
      cvt(s, P.KP + (long)b * n * cMF, cMF, P.KT16, 128, n, npad, 1);
      cvt(s, P.V_ + (long)b * n * 64 + h * 32, 64, P.VT16, 32, n, npad, 1);
      gemm(s, P.KT16, npad, P.VT16, npad, P.KV, 32, nullptr, nullptr, nullptr, 0,
           128, 32, npad, 0);
      cvt(s, P.KV, 32, P.KVT16, 32, 128, 128, 1);
      // o_h = qp @ kv
      gemm(s, P.QP16 + (long)b * n * cMF, cMF, P.KVT16, cMF,
           P.O + (long)b * n * 64 + h * 32, 64, nullptr, nullptr, nullptr, 0,
           n, 32, cMF, 0);
    }
    k_scalerows<<<gthreads(TR * 32), 256, 0, s>>>(P.O + h * 32, P.O16 + h * 32,
                                                  64, 32, P.Z, TR);
  }

  // x2 = x + o @ Wo + bo
  cvt(s, p.o.w, 256, P.W16, 256, 64, 64, 1);
  gemm(s, P.O16, 64, P.W16, 64, P.PB, 256, nullptr, p.o.b, P.PX, 256,
       (int)TR, 256, 64, 0);
  // FFN
  k_ln<<<gthreads(TR * 32), 256, 0, s>>>(P.PB, p.ln2g, p.ln2b, P.XN, P.XN16, TR, cD);
  cvt(s, p.ff1.w, 1024, P.W16, 1024, 256, 256, 1);
  gemm(s, P.XN16, 256, P.W16, 256, P.T1024, 1024, P.T1024h, p.ff1.b, nullptr, 0,
       (int)TR, 1024, 256, 2);
  cvt(s, p.ff2.w, 256, P.W16, 256, 1024, 1024, 1);
  gemm(s, P.T1024h, 1024, P.W16, 1024, P.PA, 256, nullptr, p.ff2.b, P.PB, 256,
       (int)TR, 256, 1024, 0);
  k_unpack<<<gthreads(TR * cD), 256, 0, s>>>(P.PA, base, bstride, n, nb, cD);
}

extern "C" void kernel_launch(void* const* d_in, const int* in_sizes, int n_in,
                              void* d_out, int out_size, void* d_ws, size_t ws_size,
                              hipStream_t stream) {
  (void)in_sizes; (void)n_in; (void)out_size; (void)ws_size;
  // ---- inputs (jax sorted-key flatten order) ----
  const float* x   = (const float*)d_in[0];
  const float* lat = (const float*)d_in[1];
  const int* eidx  = (const int*)d_in[2];
  const int* src = eidx, *dst = eidx + cE;
  int ii = 3;
  BlockP blocks[2];
  for (int bi = 0; bi < 2; ++bi) {
    BlockP bp;
    bp.bins = parse_perf(d_in, ii);   // stacked leaves [BINS, ...]
    bp.full = parse_perf(d_in, ii);   // P_REPEAT == 1
    bp.gat.att  = (const float*)d_in[ii++];
    bp.gat.bias = (const float*)d_in[ii++];
    bp.gat.wl.b = (const float*)d_in[ii++]; bp.gat.wl.w = (const float*)d_in[ii++];
    bp.gat.wr.b = (const float*)d_in[ii++]; bp.gat.wr.w = (const float*)d_in[ii++];
    bp.lnb = (const float*)d_in[ii++]; bp.lng = (const float*)d_in[ii++];
    bp.score.b = (const float*)d_in[ii++]; bp.score.w = (const float*)d_in[ii++];
    blocks[bi] = bp;
  }
  const float* gene_emb = (const float*)d_in[ii++];
  Dense gep1{(const float*)d_in[ii], (const float*)d_in[ii + 1]}; ii += 2;
  Dense gep2{(const float*)d_in[ii], (const float*)d_in[ii + 1]}; ii += 2;
  Dense hd1{(const float*)d_in[ii], (const float*)d_in[ii + 1]}; ii += 2;
  Dense hd2{(const float*)d_in[ii], (const float*)d_in[ii + 1]}; ii += 2;
  const float* invf = (const float*)d_in[ii++];
  const float* lnbT = (const float*)d_in[ii++];
  const float* lngT = (const float*)d_in[ii++];
  Dense xp1{(const float*)d_in[ii], (const float*)d_in[ii + 1]}; ii += 2;
  Dense xp2{(const float*)d_in[ii], (const float*)d_in[ii + 1]}; ii += 2;

  // ---- workspace bump allocator ----
  char* w = (char*)d_ws;
  size_t off = 0;
  auto alloc = [&](size_t bytes) -> void* {
    void* r = w + off;
    off = (off + bytes + 255) & ~(size_t)255;
    return r;
  };
  Pool P;
  P.H  = (float*)alloc(cBG * cD * 4);
  P.XN = (float*)alloc(cBG * cD * 4);
  P.XL = (float*)alloc(cBG * cD * 4);
  P.XR = (float*)alloc(cBG * cD * 4);
  P.GX = (float*)alloc(cBG * cD * 4);
  P.HS = (float*)alloc(cBG * cD * 4);
  P.PX = (float*)alloc(cBG * cD * 4);
  P.T1024 = (float*)alloc(cBG * 1024 * 4);
  P.Q  = (float*)alloc(cBG * 64 * 4);
  P.K_ = (float*)alloc(cBG * 64 * 4);
  P.V_ = (float*)alloc(cBG * 64 * 4);
  P.O  = (float*)alloc(cBG * 64 * 4);
  P.DD = (float*)alloc(cBG * cMF * 4);
  P.QP = (float*)alloc(cBG * cMF * 4);
  P.KP = (float*)alloc(cBG * cMF * 4);
  P.Z  = (float*)alloc(cBG * 4);
  P.KS = (float*)alloc(cB * cMF * 4);
  P.RMAX = (float*)alloc(cBG * 4);
  P.MXK  = (float*)alloc(cB * 4);
  P.EB = (float*)alloc((long)cB * cE * 4);
  P.DEN = (float*)alloc((long)cB * cG * 4);
  P.SC  = (float*)alloc(cBG * 4);
  P.KEYS = (float*)alloc((long)cB * cN2 * 4);
  P.KV = (float*)alloc(128 * 32 * 4);
  P.XN16   = (f16*)alloc(cBG * cD * 2);
  P.T1024h = (f16*)alloc(cBG * 1024 * 2);
  P.W16    = (f16*)alloc((size_t)1024 * 512 * 2);
  P.PROJ16 = (f16*)alloc(128 * 32 * 2);
  P.Q16 = (f16*)alloc(cBG * 64 * 2);
  P.K16 = (f16*)alloc(cBG * 64 * 2);
  P.O16 = (f16*)alloc(cBG * 64 * 2);
  P.QP16 = (f16*)alloc(cBG * cMF * 2);
  P.KT16 = (f16*)alloc((size_t)128 * 20032 * 2);
  P.VT16 = (f16*)alloc((size_t)32 * 20032 * 2);
  P.KVT16 = (f16*)alloc(32 * 128 * 2);
  P.MM   = (unsigned*)alloc((long)cB * cG * 4);
  P.VALS = (int*)alloc((long)cB * cN2 * 4);
  P.ORD  = (int*)alloc((long)cB * cG * 4);
  P.INV  = (int*)alloc((long)cB * cG * 4);
  P.PA = P.XR;  // free during performer phases
  P.PB = P.GX;
  float* GT = P.HS;        // gene tokens live in HS before block loop
  f16* GE16 = P.XN16;      // gene_emb f16 staging

  // ---- gene token MLP: relu(gene_emb@l1)@l2 (f16 hidden fused in epilogue) ----
  cvt(stream, gene_emb, cEMB, GE16, cG, cEMB, cEMB, 0);
  cvt(stream, gep1.w, 1024, P.W16, 1024, cEMB, cEMB, 1);
  gemm(stream, GE16, cEMB, P.W16, cEMB, P.T1024, 1024, P.T1024h, gep1.b, nullptr, 0,
       cG, 1024, cEMB, 1);
  cvt(stream, gep2.w, 256, P.W16, 256, 1024, 1024, 1);
  gemm(stream, P.T1024h, 1024, P.W16, 1024, GT, 256, nullptr, gep2.b, nullptr, 0,
       cG, 256, 1024, 0);

  // ---- token embedding + x_proj MLP ----
  k_embed<<<gthreads(cBG * cD), 256, 0, stream>>>(x, invf, GT, lat, P.H, P.XN16);
  cvt(stream, xp1.w, 1024, P.W16, 1024, 256, 256, 1);
  gemm(stream, P.XN16, 256, P.W16, 256, P.T1024, 1024, P.T1024h, xp1.b, nullptr, 0,
       (int)cBG, 1024, 256, 1);
  cvt(stream, xp2.w, 256, P.W16, 256, 1024, 1024, 1);
  gemm(stream, P.T1024h, 1024, P.W16, 1024, P.H, 256, nullptr, xp2.b, nullptr, 0,
       (int)cBG, 256, 1024, 0);

  // ---- GBFormer blocks ----
  for (int bi = 0; bi < 2; ++bi) {
    const BlockP& bp = blocks[bi];
    k_ln<<<gthreads(cBG * 32), 256, 0, stream>>>(P.H, bp.lng, bp.lnb, P.XN, P.XN16,
                                                 cBG, cD);
    cvt(stream, bp.gat.wl.w, 256, P.W16, 256, 256, 256, 1);
    gemm(stream, P.XN16, 256, P.W16, 256, P.XL, 256, nullptr, bp.gat.wl.b, nullptr, 0,
         (int)cBG, 256, 256, 0);
    cvt(stream, bp.gat.wr.w, 256, P.W16, 256, 256, 256, 1);
    gemm(stream, P.XN16, 256, P.W16, 256, P.XR, 256, nullptr, bp.gat.wr.b, nullptr, 0,
         (int)cBG, 256, 256, 0);
    k_fill_u32<<<gthreads((long)cB * cG), 256, 0, stream>>>(P.MM, 0u, (long)cB * cG);
    k_fill_f32<<<gthreads((long)cB * cG), 256, 0, stream>>>(P.DEN, 0.f, (long)cB * cG);
    k_fill_f32<<<gthreads(cBG * cD), 256, 0, stream>>>(P.GX, 0.f, cBG * (long)cD);
    k_gat_edge<<<gthreads((long)cB * cE * 32), 256, 0, stream>>>(P.XL, P.XR, src, dst,
                                                                 bp.gat.att, P.EB, P.MM);
    k_gat_soft<<<gthreads((long)cB * cE), 256, 0, stream>>>(P.EB, P.MM, dst, P.DEN);
    k_gat_agg<<<gthreads((long)cB * cE * 32), 256, 0, stream>>>(P.XL, P.EB, P.DEN,
                                                                src, dst, P.GX);
    k_gatmix<<<gthreads(cBG * cD), 256, 0, stream>>>(P.XN, P.GX, bp.gat.bias, P.XL,
                                                     cBG, cD);  // h2 -> XL
    // scores + descending argsort + inverse permutation
    k_gemv<<<gthreads(cBG * 32), 256, 0, stream>>>(P.XL, 256, 0, bp.score.w, 0,
                                                   bp.score.b, P.SC, 0, (int)cBG, 256, 0);
    k_sortinit<<<gthreads((long)cB * cN2), 256, 0, stream>>>(P.SC, P.KEYS, P.VALS);
    for (int k = 2; k <= cN2; k <<= 1)
      for (int j = k >> 1; j > 0; j >>= 1)
        k_bitonic<<<gthreads((long)cB * cN2), 256, 0, stream>>>(P.KEYS, P.VALS, k, j);
    k_copyord<<<gthreads((long)cB * cG), 256, 0, stream>>>(P.VALS, P.ORD);
    k_invperm<<<gthreads((long)cB * cG), 256, 0, stream>>>(P.ORD, P.INV);
    k_gather<<<gthreads(cBG * cD), 256, 0, stream>>>(P.XL, P.ORD, P.HS);
    // per-bin Performer experts
    for (int bin = 0; bin < cBINS; ++bin) {
      PerfP pp = perf_at_bin(bp.bins, bin);
      run_performer(stream, pp, P.HS + (long)bin * cNB * cD, (long)cG * cD,
                    cNB, cB, P);
    }
    // scatter back (take along inv) then full Performer
    k_gather<<<gthreads(cBG * cD), 256, 0, stream>>>(P.HS, P.INV, P.H);
    run_performer(stream, bp.full, P.H, (long)cG * cD, cG, cB, P);
  }

  // ---- final LN + head ----
  k_ln<<<gthreads(cBG * 32), 256, 0, stream>>>(P.H, lngT, lnbT, P.XN, P.XN16, cBG, cD);
  cvt(stream, hd1.w, 1024, P.W16, 1024, 256, 256, 1);
  gemm(stream, P.XN16, 256, P.W16, 256, P.T1024, 1024, nullptr, hd1.b, nullptr, 0,
       (int)cBG, 1024, 256, 1);
  k_gemv<<<gthreads(cBG * 32), 256, 0, stream>>>(P.T1024, 1024, 0, hd2.w, 0, hd2.b,
                                                 (float*)d_out, 0, (int)cBG, 1024, 0);
}
